// MambaModel_46222438040038
// MI455X (gfx1250) — compile-verified, matrix operand order checked by reading, not run on hardware
//
#include <hip/hip_runtime.h>
#include <math.h>

// ---- model constants (from reference) ----
#define D_IN     12
#define D_MODEL  256
#define D_STATE  16
#define D_CONV   4
#define N_LAYERS 6
#define D_OUT    128
#define D_INNER  512
#define DT_RANK  16
#define B_SZ     16
#define SEQ      2048
#define NROWS    (B_SZ * SEQ)            // 32768 token rows
#define XPROJ_N  (DT_RANK + 2 * D_STATE) // 48 real, padded to 64
#define XPROJ_NP 64
#define K_ENC_P  32                      // K=12 padded
#define K_DT_P   32                      // K=16 padded

typedef __attribute__((ext_vector_type(16))) _Float16 v16h;
typedef __attribute__((ext_vector_type(8)))  float    v8f;

// K-index mapping of the ISA 16-bit fragment layout (per lane khalf kh):
//   halves e in [0,8)  -> k = kh*8 + e
//   halves e in [8,16) -> k = 16 + kh*8 + (e-8)
__device__ __forceinline__ int kmap(int e, int kh) {
    return (e < 8) ? (kh * 8 + e) : (16 + kh * 8 + (e - 8));
}

__device__ __forceinline__ v16h cvt16(float4 a0, float4 a1, float4 a2, float4 a3) {
    v16h f;
    f[0]  = (_Float16)a0.x; f[1]  = (_Float16)a0.y;
    f[2]  = (_Float16)a0.z; f[3]  = (_Float16)a0.w;
    f[4]  = (_Float16)a1.x; f[5]  = (_Float16)a1.y;
    f[6]  = (_Float16)a1.z; f[7]  = (_Float16)a1.w;
    f[8]  = (_Float16)a2.x; f[9]  = (_Float16)a2.y;
    f[10] = (_Float16)a2.z; f[11] = (_Float16)a2.w;
    f[12] = (_Float16)a3.x; f[13] = (_Float16)a3.y;
    f[14] = (_Float16)a3.z; f[15] = (_Float16)a3.w;
    return f;
}

// =====================================================================
// Weight pre-pack: f32 row-major B[Kreal x Nreal] (ld = Nreal) ->
// fragment-ready f16: out[kb][col<Npad][kh][16 halves], zero padded.
// =====================================================================
__global__ __launch_bounds__(256)
void pack_w_k(const float* __restrict__ B, int Nreal, int Npad,
              int Kreal, int nkb, _Float16* __restrict__ out)
{
    const int idx = blockIdx.x * blockDim.x + threadIdx.x;
    const int total = nkb * Npad * 2;
    if (idx >= total) return;
    const int kh  = idx & 1;
    const int col = (idx >> 1) % Npad;
    const int kb  = (idx >> 1) / Npad;
    _Float16* dst = out + (size_t)idx * 16;
    #pragma unroll
    for (int e = 0; e < 16; ++e) {
        const int k = kb * 32 + kmap(e, kh);
        const float v = (k < Kreal && col < Nreal) ? B[(size_t)k * Nreal + col] : 0.0f;
        dst[e] = (_Float16)v;
    }
}

// ---- pad x (NROWS x 12) -> x_pad (NROWS x 32, zero-filled tail) ----
__global__ __launch_bounds__(256)
void pad_x_k(const float* __restrict__ x, float* __restrict__ xp)
{
    const int row = blockIdx.x * blockDim.x + threadIdx.x;
    if (row >= NROWS) return;
    const float* src = x + (size_t)row * D_IN;
    float* dst = xp + (size_t)row * K_ENC_P;
    #pragma unroll
    for (int i = 0; i < K_ENC_P; ++i) dst[i] = (i < D_IN) ? src[i] : 0.0f;
}

// =====================================================================
// Branch-free WMMA GEMM:  C[M x Npad] (=, +=) A[M x K] * Bp  (+bias)
// Requirements: M%256==0, Npad%64==0, K%32==0 (guaranteed by padding).
// Wave computes a 32x64 strip (2 A frags x 4 B frags = 8 WMMAs/K-step);
// block = 8 waves = 256 rows x 64 cols.
// B pointers advance by a constant stride (no per-iter 64-bit addr math).
// =====================================================================
template <int POSTOP, int ACC>
__global__ __launch_bounds__(256)
void gemm_wmma_k(const float* __restrict__ A, int lda,
                 const _Float16* __restrict__ Bp,
                 float* __restrict__ C, int ldc,
                 const float* __restrict__ bias,
                 int Npad, int K)
{
    const int lane    = threadIdx.x & 31;
    const int wave    = threadIdx.x >> 5;            // 0..7
    const int rowBase = blockIdx.y * 256 + wave * 32;
    const int colBase = blockIdx.x * 64;
    const int r       = lane & 15;
    const int kh      = lane >> 4;

    v8f acc[2][4];
    #pragma unroll
    for (int g = 0; g < 2; ++g)
        #pragma unroll
        for (int t = 0; t < 4; ++t)
            #pragma unroll
            for (int e = 0; e < 8; ++e) acc[g][t][e] = 0.0f;

    const float* arow0 = A + (size_t)(rowBase + r) * lda + kh * 8;
    const float* arow1 = arow0 + (size_t)16 * lda;

    // B fragment pointers: one per 16-col tile; +512 halves per tile,
    // +Npad*32 halves per K-step.
    const size_t  bstride = (size_t)Npad * 32;
    const _Float16* bp0 = Bp + (((size_t)(colBase + r) * 2) + kh) * 16;
    const _Float16* bp1 = bp0 + 512;
    const _Float16* bp2 = bp0 + 1024;
    const _Float16* bp3 = bp0 + 1536;

    for (int k0 = 0; k0 < K; k0 += 32) {
        // ---- issue all loads for this K-step up front ----
        const v16h bf0 = *(const v16h*)bp0; bp0 += bstride;
        const v16h bf1 = *(const v16h*)bp1; bp1 += bstride;
        const v16h bf2 = *(const v16h*)bp2; bp2 += bstride;
        const v16h bf3 = *(const v16h*)bp3; bp3 += bstride;

        const float4* ap0 = (const float4*)(arow0 + k0);
        const float4* aq0 = (const float4*)(arow0 + k0 + 16);
        const float4* ap1 = (const float4*)(arow1 + k0);
        const float4* aq1 = (const float4*)(arow1 + k0 + 16);
        const float4 x0 = ap0[0], x1 = ap0[1], x2 = aq0[0], x3 = aq0[1];
        const float4 y0 = ap1[0], y1 = ap1[1], y2 = aq1[0], y3 = aq1[1];

        // speculative prefetch of next K-step's A rows
        __builtin_prefetch(arow0 + k0 + 32, 0, 1);
        __builtin_prefetch(arow1 + k0 + 32, 0, 1);

        const v16h af0 = cvt16(x0, x1, x2, x3);
        const v16h af1 = cvt16(y0, y1, y2, y3);

        acc[0][0] = __builtin_amdgcn_wmma_f32_16x16x32_f16(false, af0, false, bf0, (short)0, acc[0][0], false, false);
        acc[0][1] = __builtin_amdgcn_wmma_f32_16x16x32_f16(false, af0, false, bf1, (short)0, acc[0][1], false, false);
        acc[0][2] = __builtin_amdgcn_wmma_f32_16x16x32_f16(false, af0, false, bf2, (short)0, acc[0][2], false, false);
        acc[0][3] = __builtin_amdgcn_wmma_f32_16x16x32_f16(false, af0, false, bf3, (short)0, acc[0][3], false, false);
        acc[1][0] = __builtin_amdgcn_wmma_f32_16x16x32_f16(false, af1, false, bf0, (short)0, acc[1][0], false, false);
        acc[1][1] = __builtin_amdgcn_wmma_f32_16x16x32_f16(false, af1, false, bf1, (short)0, acc[1][1], false, false);
        acc[1][2] = __builtin_amdgcn_wmma_f32_16x16x32_f16(false, af1, false, bf2, (short)0, acc[1][2], false, false);
        acc[1][3] = __builtin_amdgcn_wmma_f32_16x16x32_f16(false, af1, false, bf3, (short)0, acc[1][3], false, false);
    }

    // D layout: VGPR e -> row = rowBase + g*16 + e + 8*kh, col = colBase + t*16 + r
    #pragma unroll
    for (int g = 0; g < 2; ++g) {
        #pragma unroll
        for (int t = 0; t < 4; ++t) {
            const int col = colBase + t * 16 + r;
            const float bv = bias ? bias[col] : 0.0f;
            #pragma unroll
            for (int e = 0; e < 8; ++e) {
                const int row = rowBase + g * 16 + e + 8 * kh;
                float v = acc[g][t][e] + bv;
                if (POSTOP == 1) v = (v > 20.0f) ? v : log1pf(__expf(v));
                const size_t idx = (size_t)row * ldc + col;
                if (ACC) v += C[idx];
                C[idx] = v;
            }
        }
    }
}

// =====================================================================
// LayerNorm over D_MODEL=256, one wave32 per row, 8 elems/lane.
// =====================================================================
__global__ __launch_bounds__(256)
void layernorm_k(const float* __restrict__ x, const float* __restrict__ w,
                 const float* __restrict__ b, float* __restrict__ y)
{
    const int row = blockIdx.x * (blockDim.x >> 5) + (threadIdx.x >> 5);
    const int lane = threadIdx.x & 31;
    const float* xr = x + (size_t)row * D_MODEL;

    float vals[8];
    float s = 0.0f;
    #pragma unroll
    for (int i = 0; i < 8; ++i) { vals[i] = xr[lane + 32 * i]; s += vals[i]; }
    #pragma unroll
    for (int o = 16; o > 0; o >>= 1) s += __shfl_down(s, o, 32);
    s = __shfl(s, 0, 32);
    const float mu = s * (1.0f / D_MODEL);

    float v = 0.0f;
    #pragma unroll
    for (int i = 0; i < 8; ++i) { const float d = vals[i] - mu; v += d * d; }
    #pragma unroll
    for (int o = 16; o > 0; o >>= 1) v += __shfl_down(v, o, 32);
    v = __shfl(v, 0, 32);
    const float rstd = rsqrtf(v * (1.0f / D_MODEL) + 1e-5f);

    float* yr = y + (size_t)row * D_MODEL;
    #pragma unroll
    for (int i = 0; i < 8; ++i) {
        const int c = lane + 32 * i;
        yr[c] = (vals[i] - mu) * rstd * w[c] + b[c];
    }
}

// =====================================================================
// Causal depthwise conv (width 4) + bias + SiLU.
// =====================================================================
__global__ __launch_bounds__(256)
void conv_silu_k(const float* __restrict__ xz, const float* __restrict__ cw,
                 const float* __restrict__ cb, float* __restrict__ u)
{
    const size_t idx = (size_t)blockIdx.x * blockDim.x + threadIdx.x;
    if (idx >= (size_t)NROWS * D_INNER) return;
    const int d = (int)(idx % D_INNER);
    const size_t bt = idx / D_INNER;
    const int t = (int)(bt % SEQ);
    const size_t b = bt / SEQ;

    float acc = cb[d];
    #pragma unroll
    for (int j = 0; j < D_CONV; ++j) {
        const int tt = t - (D_CONV - 1) + j;
        if (tt >= 0)
            acc += xz[(b * SEQ + (size_t)tt) * (2 * D_INNER) + d] * cw[d * D_CONV + j];
    }
    u[idx] = acc / (1.0f + __expf(-acc));
}

// =====================================================================
// Selective scan. Single-wave workgroups (barriers lower to S_NOP):
// grid(B_SZ, D_INNER/32), block 32, one thread per channel.
// proj row stride is XPROJ_NP=64; B at cols [16,32), C at [32,48).
// =====================================================================
__global__ __launch_bounds__(32)
void scan_k(const float* __restrict__ u, const float* __restrict__ dt,
            const float* __restrict__ proj, const float* __restrict__ A_log,
            const float* __restrict__ Dskip, float* __restrict__ y)
{
    const int b = blockIdx.x;
    const int d = blockIdx.y * 32 + threadIdx.x;
    __shared__ float sBC[2 * D_STATE];

    float Ad[D_STATE];
    #pragma unroll
    for (int n = 0; n < D_STATE; ++n) Ad[n] = -__expf(A_log[d * D_STATE + n]);
    float hs[D_STATE];
    #pragma unroll
    for (int n = 0; n < D_STATE; ++n) hs[n] = 0.0f;
    const float Dd = Dskip[d];

    for (int t = 0; t < SEQ; ++t) {
        const size_t row = (size_t)b * SEQ + t;
        sBC[threadIdx.x] = proj[row * XPROJ_NP + DT_RANK + threadIdx.x];
        __syncthreads();   // single-wave workgroup: S_NOP, orders LDS via DScnt

        const float dtv = dt[row * D_INNER + d];
        const float uv  = u[row * D_INNER + d];
        const float du  = dtv * uv;
        float acc = 0.0f;
        #pragma unroll
        for (int n = 0; n < D_STATE; ++n) {
            hs[n] = __expf(dtv * Ad[n]) * hs[n] + du * sBC[n];
            acc  += hs[n] * sBC[D_STATE + n];
        }
        y[row * D_INNER + d] = acc + uv * Dd;
        __syncthreads();
    }
}

// =====================================================================
// Gate: y *= silu(z)
// =====================================================================
__global__ __launch_bounds__(256)
void gate_k(float* __restrict__ y, const float* __restrict__ xz)
{
    const size_t idx = (size_t)blockIdx.x * blockDim.x + threadIdx.x;
    if (idx >= (size_t)NROWS * D_INNER) return;
    const int d = (int)(idx % D_INNER);
    const size_t bt = idx / D_INNER;
    const float z = xz[bt * (2 * D_INNER) + D_INNER + d];
    y[idx] *= z / (1.0f + __expf(-z));
}

// =====================================================================
// Host launcher
// =====================================================================
static inline dim3 gemm_grid(int Npad) {
    return dim3(Npad / 64, NROWS / 256, 1);
}
static inline void launch_pack(const float* B, int Nreal, int Npad, int Kreal,
                               int Kpad, _Float16* out, hipStream_t s) {
    const int nkb = Kpad / 32;
    const int total = nkb * Npad * 2;
    pack_w_k<<<(total + 255) / 256, 256, 0, s>>>(B, Nreal, Npad, Kreal, nkb, out);
}

extern "C" void kernel_launch(void* const* d_in, const int* in_sizes, int n_in,
                              void* d_out, int out_size, void* d_ws, size_t ws_size,
                              hipStream_t stream)
{
    (void)in_sizes; (void)n_in; (void)out_size; (void)ws_size;

    const float* x       = (const float*)d_in[0];
    const float* W_enc   = (const float*)d_in[1];
    const float* ln_w    = (const float*)d_in[2];
    const float* ln_b    = (const float*)d_in[3];
    const float* W_in    = (const float*)d_in[4];
    const float* conv_w  = (const float*)d_in[5];
    const float* conv_b  = (const float*)d_in[6];
    const float* W_xproj = (const float*)d_in[7];
    const float* W_dt    = (const float*)d_in[8];
    const float* b_dt    = (const float*)d_in[9];
    const float* A_log   = (const float*)d_in[10];
    const float* D_skip  = (const float*)d_in[11];
    const float* W_out   = (const float*)d_in[12];
    const float* W_dec   = (const float*)d_in[13];
    const float* b_dec   = (const float*)d_in[14];
    float* out = (float*)d_out;

    // ---- workspace carve: f32 buffers, then 32B-aligned f16 region ----
    float* ws = (float*)d_ws;
    size_t o = 0;
    float* h     = ws + o; o += (size_t)NROWS * D_MODEL;
    float* hn    = ws + o; o += (size_t)NROWS * D_MODEL;
    float* xz    = ws + o; o += (size_t)NROWS * 2 * D_INNER;
    float* u     = ws + o; o += (size_t)NROWS * D_INNER;
    float* proj  = ws + o; o += (size_t)NROWS * XPROJ_NP;
    float* dt    = ws + o; o += (size_t)NROWS * D_INNER;
    float* y     = ws + o; o += (size_t)NROWS * D_INNER;
    float* x_pad = ws + o; o += (size_t)NROWS * K_ENC_P;

    _Float16* fp = (_Float16*)(ws + o);
    size_t fo = 0;
    const size_t szWin  = (size_t)(D_MODEL / 32) * (2 * D_INNER) * 32;
    const size_t szWxp  = (size_t)(D_INNER / 32) * XPROJ_NP * 32;
    const size_t szWdt  = (size_t)(K_DT_P / 32) * D_INNER * 32;
    const size_t szWout = (size_t)(D_INNER / 32) * D_MODEL * 32;
    _Float16* Wp_in   = fp + fo; fo += szWin  * N_LAYERS;
    _Float16* Wp_xp   = fp + fo; fo += szWxp  * N_LAYERS;
    _Float16* Wp_dt   = fp + fo; fo += szWdt  * N_LAYERS;
    _Float16* Wp_out  = fp + fo; fo += szWout * N_LAYERS;
    _Float16* Wp_enc  = fp + fo; fo += (size_t)(K_ENC_P / 32) * D_MODEL * 32;
    _Float16* Wp_dec  = fp + fo; fo += (size_t)(D_MODEL / 32) * D_OUT * 32;

    const dim3 blk(256, 1, 1);
    const size_t nElemInner = (size_t)NROWS * D_INNER;
    const dim3 gridInner((unsigned)((nElemInner + 255) / 256), 1, 1);

    // ---- pre-pack all weights to fragment-ready f16; pad x ----
    launch_pack(W_enc, D_MODEL, D_MODEL, D_IN, K_ENC_P, Wp_enc, stream);
    launch_pack(W_dec, D_OUT, D_OUT, D_MODEL, D_MODEL, Wp_dec, stream);
    for (int i = 0; i < N_LAYERS; ++i) {
        launch_pack(W_in    + (size_t)i * D_MODEL * 2 * D_INNER,
                    2 * D_INNER, 2 * D_INNER, D_MODEL, D_MODEL, Wp_in + i * szWin, stream);
        launch_pack(W_xproj + (size_t)i * D_INNER * XPROJ_N,
                    XPROJ_N, XPROJ_NP, D_INNER, D_INNER, Wp_xp + i * szWxp, stream);
        launch_pack(W_dt    + (size_t)i * DT_RANK * D_INNER,
                    D_INNER, D_INNER, DT_RANK, K_DT_P, Wp_dt + i * szWdt, stream);
        launch_pack(W_out   + (size_t)i * D_INNER * D_MODEL,
                    D_MODEL, D_MODEL, D_INNER, D_INNER, Wp_out + i * szWout, stream);
    }
    pad_x_k<<<NROWS / 256, blk, 0, stream>>>(x, x_pad);

    // ---- encoder: h = x_pad @ W_enc  (K padded to 32) ----
    gemm_wmma_k<0, 0><<<gemm_grid(D_MODEL), blk, 0, stream>>>(
        x_pad, K_ENC_P, Wp_enc, h, D_MODEL, nullptr, D_MODEL, K_ENC_P);

    for (int i = 0; i < N_LAYERS; ++i) {
        layernorm_k<<<dim3(NROWS / 8, 1, 1), blk, 0, stream>>>(
            h, ln_w + (size_t)i * D_MODEL, ln_b + (size_t)i * D_MODEL, hn);

        // xz = hn @ W_in[i]                (32768 x 256 x 1024)
        gemm_wmma_k<0, 0><<<gemm_grid(2 * D_INNER), blk, 0, stream>>>(
            hn, D_MODEL, Wp_in + i * szWin, xz, 2 * D_INNER, nullptr,
            2 * D_INNER, D_MODEL);

        conv_silu_k<<<gridInner, blk, 0, stream>>>(
            xz, conv_w + (size_t)i * D_INNER * D_CONV,
            conv_b + (size_t)i * D_INNER, u);

        // proj = u @ W_xproj[i]            (32768 x 512 x 64pad)
        gemm_wmma_k<0, 0><<<gemm_grid(XPROJ_NP), blk, 0, stream>>>(
            u, D_INNER, Wp_xp + i * szWxp, proj, XPROJ_NP, nullptr,
            XPROJ_NP, D_INNER);

        // dt = softplus(proj[:, :16] @ W_dt[i] + b_dt[i])  (K padded to 32,
        // padded weight rows are zero so proj cols 16..31 contribute nothing)
        gemm_wmma_k<1, 0><<<gemm_grid(D_INNER), blk, 0, stream>>>(
            proj, XPROJ_NP, Wp_dt + i * szWdt, dt, D_INNER,
            b_dt + (size_t)i * D_INNER, D_INNER, K_DT_P);

        scan_k<<<dim3(B_SZ, D_INNER / 32, 1), dim3(32, 1, 1), 0, stream>>>(
            u, dt, proj, A_log + (size_t)i * D_INNER * D_STATE,
            D_skip + (size_t)i * D_INNER, y);

        gate_k<<<gridInner, blk, 0, stream>>>(y, xz);

        // h += y @ W_out[i]                (residual accumulate)
        gemm_wmma_k<0, 1><<<gemm_grid(D_MODEL), blk, 0, stream>>>(
            y, D_INNER, Wp_out + i * szWout, h, D_MODEL, nullptr,
            D_MODEL, D_INNER);
    }

    // ---- decoder: out = h @ W_dec + b_dec ----
    gemm_wmma_k<0, 0><<<gemm_grid(D_OUT), blk, 0, stream>>>(
        h, D_MODEL, Wp_dec, out, D_OUT, b_dec, D_OUT, D_MODEL);
}
